// LiTrALayer_11166914969785
// MI455X (gfx1250) — matrix operand order repaired and checked by reading, not compiled, not run on hardware
//
#include <hip/hip_runtime.h>
#include <cstdint>
#include <cstddef>

constexpr int kTok = 16384;   // B*T*H*W = 2*32*16*16
constexpr int kD   = 256;
constexpr int kNH  = 8;

typedef _Float16 h8   __attribute__((ext_vector_type(8)));
typedef _Float16 v16h __attribute__((ext_vector_type(16)));
typedef float    v8f  __attribute__((ext_vector_type(8)));

// ---------------- fragment loaders (contiguous k order, A and B alike) ----------------
// A (16x32 f16): lane L holds row M=L%16; K elements contiguous: lanes<16 K=0..15,
// lanes>=16 K=16..31 => one contiguous 16-half run at hi*16.
__device__ __forceinline__ v16h ldfragA(const _Float16* __restrict__ p, int hi) {
  h8 x = *(const h8*)(p + hi * 16);
  h8 y = *(const h8*)(p + hi * 16 + 8);
  v16h r;
#pragma unroll
  for (int i = 0; i < 8; i++) { r[i] = x[i]; r[i + 8] = y[i]; }
  return r;
}
// B (32x16 f16): lane L holds col N=L%16; K elements contiguous: lanes<16 K=0..15,
// lanes>=16 K=16..31 => one contiguous 16-half run at hi*16.
__device__ __forceinline__ v16h ldfragB(const _Float16* __restrict__ p, int hi) {
  h8 x = *(const h8*)(p + hi * 16);
  h8 y = *(const h8*)(p + hi * 16 + 8);
  v16h r;
#pragma unroll
  for (int i = 0; i < 8; i++) { r[i] = x[i]; r[i + 8] = y[i]; }
  return r;
}

__device__ __forceinline__ float red16max(float x) {
#pragma unroll
  for (int m = 1; m < 16; m <<= 1) x = fmaxf(x, __shfl_xor(x, m, 32));
  return x;
}
__device__ __forceinline__ float red16sum(float x) {
#pragma unroll
  for (int m = 1; m < 16; m <<= 1) x += __shfl_xor(x, m, 32);
  return x;
}
__device__ __forceinline__ float red32sum(float x) {
#pragma unroll
  for (int m = 1; m < 32; m <<= 1) x += __shfl_xor(x, m, 32);
  return x;
}

// ---------------- weight convert + transpose: f32 [K][N] -> f16 [N][K] ----------------
__global__ void k_wcvt(const float* __restrict__ w, _Float16* __restrict__ wt, int K, int N) {
  int idx = blockIdx.x * blockDim.x + threadIdx.x;
  if (idx >= K * N) return;
  int n = idx / K;
  int k = idx - n * K;
  wt[idx] = (_Float16)w[(size_t)k * N + n];
}

// ---------------- LayerNorm over D=256, one wave per token ----------------
__global__ __launch_bounds__(256) void k_ln(const float* __restrict__ x,
                                            const float* __restrict__ g,
                                            const float* __restrict__ b,
                                            _Float16* __restrict__ out, int M) {
  int wid  = (int)((blockIdx.x * blockDim.x + threadIdx.x) >> 5);
  int lane = threadIdx.x & 31;
  if (wid >= M) return;
  const float* row = x + (size_t)wid * kD;
  float v[8], s = 0.0f, s2 = 0.0f;
#pragma unroll
  for (int i = 0; i < 8; i++) {
    v[i] = row[i * 32 + lane];
    s += v[i]; s2 += v[i] * v[i];
  }
  s = red32sum(s); s2 = red32sum(s2);
  float mu  = s * (1.0f / kD);
  float var = s2 * (1.0f / kD) - mu * mu;
  float rs  = rsqrtf(var + 1e-5f);
  _Float16* orow = out + (size_t)wid * kD;
#pragma unroll
  for (int i = 0; i < 8; i++) {
    int c = i * 32 + lane;
    orow[c] = (_Float16)((v[i] - mu) * rs * g[c] + b[c]);
  }
}

// ---------------- GEMM epilogue helper ----------------
__device__ __forceinline__ void gemm_epi(v8f acc, int mt, int hi, int col, int N,
                                         const float* __restrict__ bias,
                                         const float* __restrict__ resid,
                                         void* __restrict__ outp, int mode) {
  if (col >= N) return;   // lane-divergent OK (no WMMA after this point)
  float bv = bias ? bias[col] : 0.0f;
#pragma unroll
  for (int v = 0; v < 8; v++) {
    int rr = mt * 16 + v + hi * 8;       // C layout: lane=col, vgpr v = row (+8 hi half)
    float val = acc[v] + bv;
    if (resid) val += resid[(size_t)rr * N + col];
    if (mode == 0) {
      ((float*)outp)[(size_t)rr * N + col] = val;
    } else {
      if (mode == 2)      val = val / (1.0f + __expf(-val));
      else if (mode == 3) val = 0.5f * val * (1.0f + erff(val * 0.70710678118654752f));
      ((_Float16*)outp)[(size_t)rr * N + col] = (_Float16)val;
    }
  }
}

// ---------------- generic WMMA GEMM: out = epilogue(A[M,K] @ Bt[N,K]^T + bias (+resid)) -----
// Each wave owns a 16-row x 32-col tile: two independent accumulators => back-to-back
// v_wmma with no RAW hazard, and the A fragment is reused across both column tiles.
// mode 0: f32 out; 1: f16 out; 2: f16 SiLU; 3: f16 exact GELU
__global__ __launch_bounds__(256) void k_gemm(const _Float16* __restrict__ A,
                                              const _Float16* __restrict__ Bt,
                                              const float* __restrict__ bias,
                                              const float* __restrict__ resid,
                                              void* __restrict__ outp,
                                              int M, int N, int K, int mode) {
  int wid  = (int)((blockIdx.x * blockDim.x + threadIdx.x) >> 5);
  int lane = threadIdx.x & 31;
  int lo = lane & 15, hi = lane >> 4;
  int tilesN2 = (N + 31) >> 5;           // pairs of 16-col tiles
  int mt = wid / tilesN2;
  int np = wid - mt * tilesN2;
  if (mt * 16 >= M) return;              // wave-uniform exit: EXEC stays full

  int row  = mt * 16 + lo;
  int col0 = np * 32 + lo;
  int col1 = col0 + 16;
  const _Float16* ap = A + (size_t)row * K;

  v8f acc0 = {0, 0, 0, 0, 0, 0, 0, 0};
  v8f acc1 = {0, 0, 0, 0, 0, 0, 0, 0};

  if ((np + 1) * 32 <= N) {
    // ---- clean path: both column tiles fully in range (all GEMMs except gates N=24)
    const _Float16* bp0 = Bt + (size_t)col0 * K;
    const _Float16* bp1 = Bt + (size_t)col1 * K;
    for (int k0 = 0; k0 < K; k0 += 32) {
      v16h a  = ldfragA(ap + k0, hi);
      v16h b0 = ldfragB(bp0 + k0, hi);
      v16h b1 = ldfragB(bp1 + k0, hi);
      acc0 = __builtin_amdgcn_wmma_f32_16x16x32_f16(false, a, false, b0, (short)0, acc0, false, false);
      acc1 = __builtin_amdgcn_wmma_f32_16x16x32_f16(false, a, false, b1, (short)0, acc1, false, false);
    }
  } else {
    // ---- guarded path (tiny gates GEMM only)
    bool ok0 = col0 < N, ok1 = col1 < N;
    const _Float16* bp0 = Bt + (size_t)(ok0 ? col0 : 0) * K;
    const _Float16* bp1 = Bt + (size_t)(ok1 ? col1 : 0) * K;
    for (int k0 = 0; k0 < K; k0 += 32) {
      v16h a  = ldfragA(ap + k0, hi);
      v16h b0 = ldfragB(bp0 + k0, hi);
      v16h b1 = ldfragB(bp1 + k0, hi);
      if (!ok0) {
#pragma unroll
        for (int i = 0; i < 16; i++) b0[i] = (_Float16)0.0f;
      }
      if (!ok1) {
#pragma unroll
        for (int i = 0; i < 16; i++) b1[i] = (_Float16)0.0f;
      }
      acc0 = __builtin_amdgcn_wmma_f32_16x16x32_f16(false, a, false, b0, (short)0, acc0, false, false);
      acc1 = __builtin_amdgcn_wmma_f32_16x16x32_f16(false, a, false, b1, (short)0, acc1, false, false);
    }
  }

  gemm_epi(acc0, mt, hi, col0, N, bias, resid, outp, mode);
  gemm_epi(acc1, mt, hi, col1, N, bias, resid, outp, mode);
}

// ---------------- flash attention, one wave per (slice, head, 16-query tile) ------------
// plane 0: HW  (S=256, 64 slices, no mask)
// plane 1: HT  (S=512, 32 slices, block-causal, block=16)
// plane 2: WT  (S=512, 32 slices, block-causal, block=16)
__global__ __launch_bounds__(256) void k_attn(const _Float16* __restrict__ qh,
                                              const _Float16* __restrict__ kh,
                                              const _Float16* __restrict__ vh,
                                              _Float16* __restrict__ oh, int plane) {
  const int S      = (plane == 0) ? 256 : 512;
  const int numS   = (plane == 0) ? 64 : 32;
  const int qtiles = S >> 4;

  int wid  = (int)((blockIdx.x * blockDim.x + threadIdx.x) >> 5);
  int lane = threadIdx.x & 31;
  int lo = lane & 15, hi = lane >> 4;

  int per = kNH * qtiles;
  int s = wid / per;
  if (s >= numS) return;                 // wave-uniform
  int r    = wid - s * per;
  int head = r / qtiles;
  int qt   = r - head * qtiles;
  int hoff = head * 32;

  // token(j) = base + (j>>4)*sT + (j&15)*sU
  int base, sT, sU;
  if (plane == 0)      { base = s * 256;                          sT = 16;  sU = 1;  }
  else if (plane == 1) { base = (s >> 4) * 8192 + (s & 15);       sT = 256; sU = 16; }
  else                 { base = (s >> 4) * 8192 + (s & 15) * 16;  sT = 256; sU = 1;  }
  const int limit = (plane == 0) ? S : (qt + 1) * 16;   // tile-granular causal bound

  __shared__ __align__(16) _Float16 psh[8][512];        // 1KB P-tile per wave
  _Float16* pl = &psh[threadIdx.x >> 5][0];

  int qj = qt * 16 + lo;
  const _Float16* qp = qh + (size_t)(base + (qj >> 4) * sT + (qj & 15) * sU) * kD + hoff;
  v16h qa = ldfragA(qp, hi);

  float mrow[8], lrow[8];
  v8f o0 = {0, 0, 0, 0, 0, 0, 0, 0};
  v8f o1 = {0, 0, 0, 0, 0, 0, 0, 0};
#pragma unroll
  for (int v = 0; v < 8; v++) { mrow[v] = -3.0e38f; lrow[v] = 0.0f; }

  const float scale = 0.17677669529663687f;   // 1/sqrt(32)
  const v8f zero = {0, 0, 0, 0, 0, 0, 0, 0};

  for (int k0 = 0; k0 < limit; k0 += 32) {
    int nk = limit - k0; if (nk > 32) nk = 32;
    bool two = (nk > 16);                // wave-uniform

    int kj0 = k0 + lo;
    const _Float16* kp0 = kh + (size_t)(base + (kj0 >> 4) * sT + (kj0 & 15) * sU) * kD + hoff;
    v16h kf0 = ldfragB(kp0, hi);
    v8f s0 = __builtin_amdgcn_wmma_f32_16x16x32_f16(false, qa, false, kf0, (short)0, zero, false, false);

    v8f s1 = zero;
    if (two) {
      int kj1 = k0 + 16 + lo;
      const _Float16* kp1 = kh + (size_t)(base + (kj1 >> 4) * sT + (kj1 & 15) * sU) * kD + hoff;
      v16h kf1 = ldfragB(kp1, hi);
      s1 = __builtin_amdgcn_wmma_f32_16x16x32_f16(false, qa, false, kf1, (short)0, zero, false, false);
    }

    // streaming softmax update (row r lives across 16 lanes at vgpr v, r = v + hi*8)
#pragma unroll
    for (int v = 0; v < 8; v++) {
      float t = s0[v] * scale;
      if (two) t = fmaxf(t, s1[v] * scale);
      t = red16max(t);
      float mn = fmaxf(mrow[v], t);
      float f  = __expf(mrow[v] - mn);
      mrow[v] = mn;
      int prow = v + hi * 8;
      float p0 = __expf(s0[v] * scale - mn);
      float p1 = two ? __expf(s1[v] * scale - mn) : 0.0f;   // masked half -> 0
      pl[prow * 32 + lo]      = (_Float16)p0;
      pl[prow * 32 + 16 + lo] = (_Float16)p1;
      float rs = red16sum(p0 + p1);
      lrow[v] = lrow[v] * f + rs;
      o0[v] *= f;
      o1[v] *= f;
    }

    asm volatile("s_wait_dscnt 0" ::: "memory");  // P stores visible before re-fragment

    v16h pa = ldfragA(pl + lo * 32, hi);          // P as A (16x32)

    v16h vb0, vb1;                                 // V as B: gather dims per lane
#pragma unroll
    for (int e = 0; e < 16; e++) {
      int kj = k0 + hi * 16 + e;                   // garbage rows killed by P==0
      size_t tb = (size_t)(base + (kj >> 4) * sT + (kj & 15) * sU) * kD + hoff;
      vb0[e] = vh[tb + lo];
      vb1[e] = vh[tb + 16 + lo];
    }
    o0 = __builtin_amdgcn_wmma_f32_16x16x32_f16(false, pa, false, vb0, (short)0, o0, false, false);
    o1 = __builtin_amdgcn_wmma_f32_16x16x32_f16(false, pa, false, vb1, (short)0, o1, false, false);
  }

#pragma unroll
  for (int v = 0; v < 8; v++) {
    int rj = qt * 16 + v + hi * 8;
    float inv = 1.0f / lrow[v];
    size_t ob = (size_t)(base + (rj >> 4) * sT + (rj & 15) * sU) * kD + hoff;
    oh[ob + lo]      = (_Float16)(o0[v] * inv);
    oh[ob + 16 + lo] = (_Float16)(o1[v] * inv);
  }
}

// ---------------- per-head 3-way gate softmax + plane mix (in-place into o_hw) ----------
__global__ __launch_bounds__(256) void k_combine(const float* __restrict__ gl,
                                                 _Float16* __restrict__ ohw,
                                                 const _Float16* __restrict__ oht,
                                                 const _Float16* __restrict__ owt) {
  int idx = blockIdx.x * blockDim.x + threadIdx.x;
  if (idx >= kTok * kD) return;
  int tokn = idx >> 8;
  int head = (idx & 255) >> 5;
  const float* g = gl + (size_t)tokn * 24 + head * 3;
  float l0 = g[0], l1 = g[1], l2 = g[2];
  float mx = fmaxf(l0, fmaxf(l1, l2));
  float e0 = __expf(l0 - mx), e1 = __expf(l1 - mx), e2 = __expf(l2 - mx);
  float inv = 1.0f / (e0 + e1 + e2);
  float m = (e0 * (float)ohw[idx] + e1 * (float)oht[idx] + e2 * (float)owt[idx]) * inv;
  ohw[idx] = (_Float16)m;
}

// ============================== host orchestration ==============================
extern "C" void kernel_launch(void* const* d_in, const int* in_sizes, int n_in,
                              void* d_out, int out_size, void* d_ws, size_t ws_size,
                              hipStream_t stream) {
  (void)in_sizes; (void)n_in; (void)out_size; (void)ws_size;
  char* ws = (char*)d_ws;
  size_t off = 0;
  auto alloc = [&](size_t bytes) { size_t r = off; off += (bytes + 255) & ~(size_t)255; return r; };

  const size_t szDD  = (size_t)kD * kD * sizeof(_Float16);
  const size_t szAct = (size_t)kTok * kD * sizeof(_Float16);

  size_t oWqkv[9];
  for (int i = 0; i < 9; i++) oWqkv[i] = alloc(szDD);
  size_t oWo  = alloc(szDD);
  size_t oWg1 = alloc(szDD);
  size_t oWg2 = alloc((size_t)24 * kD * sizeof(_Float16));
  size_t oWm1 = alloc((size_t)kD * 1024 * sizeof(_Float16));
  size_t oWm2 = alloc((size_t)kD * 1024 * sizeof(_Float16));
  size_t oYh  = alloc(szAct);
  size_t oGL  = alloc((size_t)kTok * 24 * sizeof(float));
  size_t oQ   = alloc(szAct);
  size_t oK   = alloc(szAct);            // K,V contiguous -> reused as f32 x1
  size_t oV   = alloc(szAct); (void)oV;
  size_t oOhw = alloc(szAct);
  size_t oOht = alloc(szAct);
  size_t oOwt = alloc(szAct);
  size_t oH   = alloc((size_t)kTok * 1024 * sizeof(_Float16));
  size_t oX1  = oK;                      // 16 MB f32 aliases K+V (dead after attention)
  size_t oY2  = oOht;                    // ln2 output aliases o_ht (dead after combine)

  auto wcvt = [&](int src, size_t dst, int Kd, int Nd) {
    int total = Kd * Nd;
    k_wcvt<<<dim3((total + 255) / 256), dim3(256), 0, stream>>>(
        (const float*)d_in[src], (_Float16*)(ws + dst), Kd, Nd);
  };
  for (int p = 0; p < 9; p++) wcvt(3 + 2 * p, oWqkv[p], kD, kD);
  wcvt(21, oWo, kD, kD);
  wcvt(23, oWg1, kD, kD);
  wcvt(25, oWg2, kD, 24);
  wcvt(29, oWm1, kD, 1024);
  wcvt(31, oWm2, 1024, kD);

  auto gemm = [&](const void* A, size_t bOff, const float* bias, const float* resid,
                  void* outp, int N, int K, int mode) {
    int tilesN2 = (N + 31) / 32;
    long waves = (long)(kTok / 16) * tilesN2;
    unsigned blocks = (unsigned)((waves * 32 + 255) / 256);
    k_gemm<<<dim3(blocks), dim3(256), 0, stream>>>(
        (const _Float16*)A, (const _Float16*)(ws + bOff), bias, resid, outp, kTok, N, K, mode);
  };

  const float* xin = (const float*)d_in[0];

  // 1) y = LN(x)
  k_ln<<<dim3(kTok / 8), dim3(256), 0, stream>>>(
      xin, (const float*)d_in[1], (const float*)d_in[2], (_Float16*)(ws + oYh), kTok);

  // 2) gate path first (g1 borrows Q buffer, freed before projections)
  gemm(ws + oYh, oWg1, (const float*)d_in[24], nullptr, ws + oQ, kD, kD, 2);   // silu
  gemm(ws + oQ,  oWg2, (const float*)d_in[26], nullptr, ws + oGL, 24, kD, 0);  // logits f32

  // 3) three attention planes
  size_t oO[3] = {oOhw, oOht, oOwt};
  for (int p = 0; p < 3; p++) {
    gemm(ws + oYh, oWqkv[3 * p + 0], (const float*)d_in[4 + 6 * p], nullptr, ws + oQ, kD, kD, 1);
    gemm(ws + oYh, oWqkv[3 * p + 1], (const float*)d_in[6 + 6 * p], nullptr, ws + oK, kD, kD, 1);
    gemm(ws + oYh, oWqkv[3 * p + 2], (const float*)d_in[8 + 6 * p], nullptr, ws + oV, kD, kD, 1);
    k_attn<<<dim3(1024), dim3(256), 0, stream>>>(
        (const _Float16*)(ws + oQ), (const _Float16*)(ws + oK),
        (const _Float16*)(ws + oV), (_Float16*)(ws + oO[p]), p);
  }

  // 4) gated fusion (in-place into o_hw => m)
  k_combine<<<dim3(kTok * kD / 256), dim3(256), 0, stream>>>(
      (const float*)(ws + oGL), (_Float16*)(ws + oOhw),
      (const _Float16*)(ws + oOht), (const _Float16*)(ws + oOwt));

  // 5) x1 = x + m @ wo + bo   (f32)
  gemm(ws + oOhw, oWo, (const float*)d_in[22], xin, ws + oX1, kD, kD, 0);

  // 6) MLP: out = x1 + gelu(LN(x1) @ wm1 + bm1) @ wm2 + bm2
  k_ln<<<dim3(kTok / 8), dim3(256), 0, stream>>>(
      (const float*)(ws + oX1), (const float*)d_in[27], (const float*)d_in[28],
      (_Float16*)(ws + oY2), kTok);
  gemm(ws + oY2, oWm1, (const float*)d_in[30], nullptr, ws + oH, 1024, kD, 3);           // gelu
  gemm(ws + oH,  oWm2, (const float*)d_in[32], (const float*)(ws + oX1), d_out, kD, 1024, 0);
}
